// CNNDLGA_89069031785064
// MI455X (gfx1250) — matrix operand order repaired and checked by reading, not compiled, not run on hardware
//
#include <hip/hip_runtime.h>
#include <hip/hip_bf16.h>
#include <cstdint>

typedef __attribute__((ext_vector_type(16))) _Float16 v16h;
typedef __attribute__((ext_vector_type(8)))  _Float16 v8h;
typedef __attribute__((ext_vector_type(8)))  float    v8f;

#define NB 16
#define NS 8
#define ND 64
#define N0F 40
#define NCLSF 10
#define NNUMF 30
#define NVOC 100
#define NTOT 264
#define FIN 135168            // NS*NTOT*ND
#define NH 256
#define NCLS2 19
#define KSPLIT 66
#define KCHUNK 2048           // FIN / KSPLIT

// ---------------- 1. embedding + layernorm -> x0 (B,S,40,64) f32 -----------

__global__ void embed_ln_kernel(const int* xc, const float* xn, const float* cls_emb,
                                const float* num_emb, const float* lnw, const float* lnb,
                                float* x0) {
  int n  = blockIdx.x % N0F;
  int bs = blockIdx.x / N0F;            // b*NS + s
  int d  = threadIdx.x;                 // 64 threads
  float v;
  if (n < NCLSF) {
    int tok = xc[bs * NCLSF + n];
    v = cls_emb[((size_t)n * NVOC + tok) * ND + d];
  } else {
    v = xn[bs * NNUMF + (n - NCLSF)] * num_emb[(n - NCLSF) * ND + d];
  }
  __shared__ float sm[ND];
  sm[d] = v; __syncthreads();
  for (int s = 32; s > 0; s >>= 1) { if (d < s) sm[d] += sm[d + s]; __syncthreads(); }
  float mean = sm[0] * (1.f / ND); __syncthreads();
  float c = v - mean;
  sm[d] = c * c; __syncthreads();
  for (int s = 32; s > 0; s >>= 1) { if (d < s) sm[d] += sm[d + s]; __syncthreads(); }
  float var = sm[0] * (1.f / ND);
  x0[(size_t)(bs * N0F + n) * ND + d] = lnw[d] * c * rsqrtf(var + 1e-12f) + lnb[d];
}

// ---------------- 2a. fold over S: hf = sum_s ch_w[s]*hi, xf = sum_s cx_w[s]*x0

__global__ void fold_kernel(const float* hi, const float* x0, const float* chw,
                            const float* cxw, float* hf, float* xf, int rf) {
  int idx = blockIdx.x * blockDim.x + threadIdx.x;
  int rows = rf + N0F;
  if (idx >= NB * rows * ND) return;
  int d = idx % ND;
  int row = (idx / ND) % rows;
  int b = idx / (ND * rows);
  if (row < rf) {
    float s = 0.f;
    for (int i = 0; i < NS; ++i)
      s += chw[i] * hi[(size_t)((b * NS + i) * rf + row) * ND + d];
    hf[(size_t)(b * rf + row) * ND + d] = s;
  } else {
    int n = row - rf;
    float s = 0.f;
    for (int i = 0; i < NS; ++i)
      s += cxw[i] * x0[(size_t)((b * NS + i) * N0F + n) * ND + d];
    xf[(size_t)(b * N0F + n) * ND + d] = s;
  }
}

// ---------------- 2b. q = xf@q_w^T (B,40,32), k = hf@k_w^T (B,rf,32) --------

__global__ void qk_kernel(const float* xf, const float* hf, const float* qw,
                          const float* kw, float* qb, float* kb, int rf) {
  int idx = blockIdx.x * blockDim.x + threadIdx.x;
  int rows = N0F + rf;
  if (idx >= NB * rows * 32) return;
  int e = idx & 31;
  int row = (idx >> 5) % rows;
  int b = idx / (rows * 32);
  if (row < N0F) {
    const float* x = xf + (size_t)(b * N0F + row) * ND;
    const float* w = qw + (size_t)e * ND;
    float s = 0.f;
    for (int d = 0; d < ND; ++d) s += x[d] * w[d];
    qb[(size_t)(b * N0F + row) * 32 + e] = s;
  } else {
    int r = row - N0F;
    const float* x = hf + (size_t)(b * rf + r) * ND;
    const float* w = kw + (size_t)e * ND;
    float s = 0.f;
    for (int d = 0; d < ND; ++d) s += x[d] * w[d];
    kb[(size_t)(b * rf + r) * 32 + e] = s;
  }
}

// ---------------- 2c. att softmax + fold modulation into f16 weights --------

__global__ void att_w_kernel(const float* qb, const float* kb, const float* gp,
                             const float* pfs, _Float16* W16, int rf, int OUT) {
  int n = blockIdx.x, b = blockIdx.y;
  int r = threadIdx.x;                  // 128 threads
  __shared__ float sm[128];
  __shared__ float attS[128];
  float lg = -1e30f;
  if (r < rf) {
    const float* q = qb + (size_t)(b * N0F + n) * 32;
    const float* k = kb + (size_t)(b * rf + r) * 32;
    float d = 0.f;
    for (int e = 0; e < 32; ++e) d += q[e] * k[e];
    lg = d;
  }
  sm[r] = lg; __syncthreads();
  for (int s = 64; s > 0; s >>= 1) { if (r < s) sm[r] = fmaxf(sm[r], sm[r + s]); __syncthreads(); }
  float MX = sm[0]; __syncthreads();
  float ex = (r < rf) ? __expf(lg - MX) : 0.f;
  sm[r] = ex; __syncthreads();
  for (int s = 64; s > 0; s >>= 1) { if (r < s) sm[r] += sm[r + s]; __syncthreads(); }
  float SUM = sm[0]; __syncthreads();
  attS[r] = ex / SUM;
  __syncthreads();
  float g = gp[0];
  int K = N0F * rf;
  for (int idx = r; idx < OUT * rf; idx += 128) {
    int o = idx / rf, rr = idx - o * rf;
    float wv = pfs[(size_t)o * K + n * rf + rr];
    W16[((size_t)b * OUT + o) * K + n * rf + rr] = (_Float16)(wv * (1.f + g * attS[rr]));
  }
}

// ---------------- 2d. cross GEMM: C_b(o x 512) = W_b(o x K) @ V_b(K x 512) --
// LDS tiles stored pre-swizzled in WMMA fragment layout:
//   tile[slot = tileIdx*32 + lane][e=0..15] -> one contiguous 32B v16h per lane.
// All 8 B fragments are hoisted ahead of the WMMA chain so ds_load_b128s
// overlap and the 8 v_wmma issue back-to-back.

__global__ void cross_gemm_kernel(const float* x0, const float* hi, const _Float16* W16,
                                  float* out, int rf, int OUT) {
  __shared__ _Float16 Aswz[8 * 32 * 16];   // [mt][lane][e], 8KB
  __shared__ _Float16 Vswz[8 * 32 * 16];   // [nt][lane][e], 8KB
  int b  = blockIdx.y;
  int c0 = blockIdx.x * 128;
  int tid = threadIdx.x, nth = blockDim.x;
  int wave = tid >> 5, lane = tid & 31;
  int K = N0F * rf;                        // 1600 / 5120 / 2560 (all % 32 == 0)
  v8f acc[8] = {};
  const _Float16* Wb = W16 + (size_t)b * OUT * K;
  for (int k0 = 0; k0 < K; k0 += 32) {
    // ---- A tile: swizzled copy, 8-f16 chunks stay contiguous ----
    for (int i = tid; i < OUT * 4; i += nth) {
      int row = i >> 2, c = i & 3;                   // c = K-chunk of 8
      v8h src = *(const v8h*)&Wb[(size_t)row * K + k0 + c * 8];
      int mt = row >> 4, M = row & 15;
      int half = c >> 1, ebase = 8 * (c & 1);        // A: K = e + 16*half
      *(v8h*)&Aswz[(mt * 32 + 16 * half + M) * 16 + ebase] = src;
    }
    // ---- V tile: each thread builds one lane-fragment (16 products) ----
    for (int slot = tid; slot < 256; slot += nth) {  // 8 nt * 32 lanes
      int nt = slot >> 5, l = slot & 31;
      int half = l >> 4, N = l & 15;
      int c = c0 + nt * 16 + N, s = c >> 6, d = c & 63;
      int k = k0 + 16 * half;                        // B: K = e + 16*half
      int n = k / rf, r = k - n * rf;
      const float* xbase = x0 + (size_t)(b * NS + s) * N0F * ND + d;
      const float* hbase = hi + (size_t)(b * NS + s) * rf * ND + d;
      v16h vv;
#pragma unroll
      for (int e = 0; e < 16; ++e) {
        vv[e] = (_Float16)(xbase[n * ND] * hbase[r * ND]);
        ++r; if (r == rf) { r = 0; ++n; }
      }
      *(v16h*)&Vswz[slot * 16] = vv;
    }
    __syncthreads();
    v16h af = *(const v16h*)&Aswz[(wave * 32 + lane) * 16];
    v16h bf[8];
#pragma unroll
    for (int nt = 0; nt < 8; ++nt)
      bf[nt] = *(const v16h*)&Vswz[(nt * 32 + lane) * 16];
#pragma unroll
    for (int nt = 0; nt < 8; ++nt)
      acc[nt] = __builtin_amdgcn_wmma_f32_16x16x32_f16(
          false, af, false, bf[nt], (short)0, acc[nt], false, false);
    __syncthreads();
  }
  int half = lane >> 4, Nn = lane & 15;
#pragma unroll
  for (int nt = 0; nt < 8; ++nt) {
#pragma unroll
    for (int v = 0; v < 8; ++v) {
      int orow = wave * 16 + v + 8 * half;           // C: M = v + 8*half
      int c = c0 + nt * 16 + Nn, s = c >> 6, d = c & 63;
      out[(size_t)((b * NS + s) * OUT + orow) * ND + d] = acc[nt][v];
    }
  }
}

// ---------------- 3. field-attention scores + softmax + modulated f16 flatten

__device__ inline const float* Xrow(const float* h1, const float* h2, const float* h3,
                                    const float* x0, int b, int s, int f) {
  int bs = b * NS + s;
  if (f < 128) return h1 + (size_t)(bs * 128 + f) * ND;
  if (f < 192) return h2 + (size_t)(bs * 64 + (f - 128)) * ND;
  if (f < 224) return h3 + (size_t)(bs * 32 + (f - 192)) * ND;
  return x0 + (size_t)(bs * N0F + (f - 224)) * ND;
}

__global__ void scores_kernel(const float* h1, const float* h2, const float* h3,
                              const float* x0, const float* fa_w, float* scores) {
  int f = blockIdx.x % NTOT;
  int b = blockIdx.x / NTOT;
  int e = threadIdx.x;                  // 64 threads
  float acc[NS] = {};
  for (int i = 0; i < NS; ++i) {
    float xv = Xrow(h1, h2, h3, x0, b, i, f)[e];
#pragma unroll
    for (int o = 0; o < NS; ++o) acc[o] += xv * fa_w[(size_t)(o * NS + i) * ND + e];
  }
  __shared__ float sm[ND];
  for (int o = 0; o < NS; ++o) {
    sm[e] = acc[o]; __syncthreads();
    for (int s = 32; s > 0; s >>= 1) { if (e < s) sm[e] += sm[e + s]; __syncthreads(); }
    if (e == 0) scores[(size_t)(b * NS + o) * NTOT + f] = sm[0];
    __syncthreads();
  }
}

__global__ void softmax_mod_kernel(const float* h1, const float* h2, const float* h3,
                                   const float* x0, const float* scores,
                                   const float* gammap, _Float16* flatX) {
  int b = blockIdx.x;
  int t = threadIdx.x;                  // 256 threads
  __shared__ float sm[256];
  const float* sc = scores + (size_t)b * NS * NTOT;
  float mx = -1e30f;
  for (int i = t; i < NS * NTOT; i += 256) mx = fmaxf(mx, sc[i]);
  sm[t] = mx; __syncthreads();
  for (int s = 128; s > 0; s >>= 1) { if (t < s) sm[t] = fmaxf(sm[t], sm[t + s]); __syncthreads(); }
  float MX = sm[0]; __syncthreads();
  float sum = 0.f;
  for (int i = t; i < NS * NTOT; i += 256) sum += __expf(sc[i] - MX);
  sm[t] = sum; __syncthreads();
  for (int s = 128; s > 0; s >>= 1) { if (t < s) sm[t] += sm[t + s]; __syncthreads(); }
  float inv = 1.f / sm[0];
  float gamma = gammap[0];
  for (int i = t; i < FIN; i += 256) {
    int d = i & 63;
    int f = (i >> 6) % NTOT;
    int s = i / (NTOT * ND);
    float p = __expf(sc[s * NTOT + f] - MX) * inv;
    float xv = Xrow(h1, h2, h3, x0, b, s, f)[d];
    flatX[(size_t)b * FIN + i] = (_Float16)(xv * (1.f + gamma * p));
  }
}

// ---------------- 4. fc1 WMMA GEMM (16 x 256 x 135168), K split 66 ways -----
// Memory-bound on fc1_w (138MB @ 23.3TB/s); 1056 waves stream it once.

__global__ __launch_bounds__(32) void fc1_gemm_kernel(const _Float16* flatX,
                                                      const float* w, float* hpart) {
  int nt = blockIdx.x;                  // N tile 0..15
  int kc = blockIdx.y;                  // K chunk 0..65
  int lane = threadIdx.x & 31;
  int L16 = lane & 15, half = lane >> 4;
  int n0 = nt * 16;
  v8f acc = {};
  const _Float16* arow = flatX + (size_t)L16 * FIN;      // A: M = lane&15 (batch)
  const float* brow = w + (size_t)(n0 + L16) * FIN;      // B col N = fc1_w row
  for (int step = 0; step < KCHUNK / 32; ++step) {
    int k0 = kc * KCHUNK + step * 32;
    // A fragment: two contiguous 16B chunks (K = e+16h and K = e+8+16h)
    const _Float16* ap = arow + k0 + 16 * half;
    v8h alo = *(const v8h*)ap;
    v8h ahi = *(const v8h*)(ap + 8);
    v16h a;
#pragma unroll
    for (int e = 0; e < 8; ++e) { a[e] = alo[e]; a[e + 8] = ahi[e]; }
    // B fragment: 64B contiguous f32 per lane, convert to f16
    const float* bp = brow + k0 + 16 * half;
    __builtin_prefetch(bp + 256, 0, 1);                  // global_prefetch ahead
    float4 b0 = *(const float4*)(bp);
    float4 b1 = *(const float4*)(bp + 4);
    float4 b2 = *(const float4*)(bp + 8);
    float4 b3 = *(const float4*)(bp + 12);
    v16h bb;
    bb[0]  = (_Float16)b0.x; bb[1]  = (_Float16)b0.y; bb[2]  = (_Float16)b0.z; bb[3]  = (_Float16)b0.w;
    bb[4]  = (_Float16)b1.x; bb[5]  = (_Float16)b1.y; bb[6]  = (_Float16)b1.z; bb[7]  = (_Float16)b1.w;
    bb[8]  = (_Float16)b2.x; bb[9]  = (_Float16)b2.y; bb[10] = (_Float16)b2.z; bb[11] = (_Float16)b2.w;
    bb[12] = (_Float16)b3.x; bb[13] = (_Float16)b3.y; bb[14] = (_Float16)b3.z; bb[15] = (_Float16)b3.w;
    acc = __builtin_amdgcn_wmma_f32_16x16x32_f16(
        false, a, false, bb, (short)0, acc, false, false);
  }
#pragma unroll
  for (int v = 0; v < 8; ++v) {
    int m = v + 8 * half;
    hpart[(size_t)(kc * NB + m) * NH + n0 + L16] = acc[v];
  }
}

__global__ void reduce_prelu_kernel(const float* hpart, const float* fc1b,
                                    const float* pa, float* hact) {
  int b = blockIdx.x, j = threadIdx.x;  // 16 x 256
  float s = fc1b[j];
  for (int kc = 0; kc < KSPLIT; ++kc) s += hpart[(size_t)(kc * NB + b) * NH + j];
  float a = pa[0];
  hact[b * NH + j] = (s >= 0.f) ? s : a * s;
}

__global__ void fc2_kernel(const float* hact, const float* w2, const float* b2, float* out) {
  int idx = blockIdx.x * blockDim.x + threadIdx.x;
  if (idx >= NB * NCLS2) return;
  int b = idx / NCLS2, c = idx % NCLS2;
  float s = b2[c];
  for (int j = 0; j < NH; ++j) s += hact[b * NH + j] * w2[c * NH + j];
  out[idx] = s;
}

// ---------------- launch ----------------------------------------------------

extern "C" void kernel_launch(void* const* d_in, const int* in_sizes, int n_in,
                              void* d_out, int out_size, void* d_ws, size_t ws_size,
                              hipStream_t stream) {
  (void)in_sizes; (void)n_in; (void)out_size; (void)ws_size;
  const int*   xc      = (const int*)d_in[0];
  const float* xn      = (const float*)d_in[1];
  const float* cls_emb = (const float*)d_in[2];
  const float* num_emb = (const float*)d_in[3];
  const float* lnw     = (const float*)d_in[4];
  const float* lnb     = (const float*)d_in[5];
  const float* fa_w    = (const float*)d_in[24];
  const float* gammap  = (const float*)d_in[25];
  const float* fc1w    = (const float*)d_in[26];
  const float* fc1b    = (const float*)d_in[27];
  const float* pa      = (const float*)d_in[28];
  const float* fc2w    = (const float*)d_in[29];
  const float* fc2b    = (const float*)d_in[30];

  char* ws = (char*)d_ws;
  size_t off = 0;
  auto alloc = [&](size_t bytes) -> char* {
    char* p = ws + off;
    off = (off + bytes + 255) & ~(size_t)255;
    return p;
  };
  float*    x0    = (float*)alloc((size_t)NB * NS * N0F * ND * 4);
  float*    h1    = (float*)alloc((size_t)NB * NS * 128 * ND * 4);
  float*    h2    = (float*)alloc((size_t)NB * NS * 64 * ND * 4);
  float*    h3    = (float*)alloc((size_t)NB * NS * 32 * ND * 4);
  float*    hf    = (float*)alloc((size_t)NB * 128 * ND * 4);
  float*    xf    = (float*)alloc((size_t)NB * N0F * ND * 4);
  float*    qb    = (float*)alloc((size_t)NB * N0F * 32 * 4);
  float*    kb    = (float*)alloc((size_t)NB * 128 * 32 * 4);
  _Float16* W16   = (_Float16*)alloc((size_t)NB * 64 * 5120 * 2);  // max layer
  float*    sc    = (float*)alloc((size_t)NB * NS * NTOT * 4);
  _Float16* flatX = (_Float16*)alloc((size_t)NB * FIN * 2);
  float*    hpart = (float*)alloc((size_t)KSPLIT * NB * NH * 4);
  float*    hact  = (float*)alloc((size_t)NB * NH * 4);

  embed_ln_kernel<<<NB * NS * N0F, ND, 0, stream>>>(xc, xn, cls_emb, num_emb, lnw, lnb, x0);

  const int rfs[3]  = {40, 128, 64};
  const int outs[3] = {128, 64, 32};
  float* houts[3]   = {h1, h2, h3};
  const float* hcur = x0;
  for (int L = 0; L < 3; ++L) {
    const float* chw = (const float*)d_in[6 + 6 * L + 0];
    const float* cxw = (const float*)d_in[6 + 6 * L + 1];
    const float* qw  = (const float*)d_in[6 + 6 * L + 2];
    const float* kw  = (const float*)d_in[6 + 6 * L + 3];
    const float* g   = (const float*)d_in[6 + 6 * L + 4];
    const float* pfs = (const float*)d_in[6 + 6 * L + 5];
    int rf = rfs[L], OUT = outs[L];

    int tf = NB * (rf + N0F) * ND;
    fold_kernel<<<(tf + 255) / 256, 256, 0, stream>>>(hcur, x0, chw, cxw, hf, xf, rf);
    int tq = NB * (N0F + rf) * 32;
    qk_kernel<<<(tq + 255) / 256, 256, 0, stream>>>(xf, hf, qw, kw, qb, kb, rf);
    att_w_kernel<<<dim3(N0F, NB), 128, 0, stream>>>(qb, kb, g, pfs, W16, rf, OUT);
    cross_gemm_kernel<<<dim3(4, NB), (OUT / 16) * 32, 0, stream>>>(x0, hcur, W16,
                                                                   houts[L], rf, OUT);
    hcur = houts[L];
  }

  scores_kernel<<<NB * NTOT, ND, 0, stream>>>(h1, h2, h3, x0, fa_w, sc);
  softmax_mod_kernel<<<NB, 256, 0, stream>>>(h1, h2, h3, x0, sc, gammap, flatX);
  fc1_gemm_kernel<<<dim3(16, KSPLIT), 32, 0, stream>>>(flatX, fc1w, hpart);
  reduce_prelu_kernel<<<NB, NH, 0, stream>>>(hpart, fc1b, pa, hact);
  fc2_kernel<<<1, 320, 0, stream>>>(hact, fc2w, fc2b, (float*)d_out);
}
